// SelfAttention_61856118997165
// MI455X (gfx1250) — compile-verified, matrix operand order checked by reading, not run on hardware
//
#include <hip/hip_runtime.h>
#include <hip/hip_bf16.h>
#include <math.h>

#define BB 4
#define LL 2048
#define HH 1024
#define NHEADS 16
#define DD 64
#define BL (BB * LL)   // 8192 token rows

typedef __attribute__((ext_vector_type(16))) _Float16 v16h;
typedef __attribute__((ext_vector_type(8)))  _Float16 v8h;
typedef __attribute__((ext_vector_type(8)))  float    v8f;

__device__ __forceinline__ v8f wmma_f32_f16(v16h a, v16h b, v8f c) {
  // D = A(16x32 f16) * B(32x16 f16) + C(16x16 f32)
  return __builtin_amdgcn_wmma_f32_16x16x32_f16(false, a, false, b, (short)0, c,
                                                false, false);
}

// Load a 16x32 f16 fragment (A layout; identical pattern serves as B^T rows)
// from a row-major f16 matrix with leading dim `ld`, tile origin (row0, k0).
// Per ISA 7.12.2: lane<16 holds K = {k0..k0+7, k0+16..k0+23} of row (row0+lane),
// lane>=16 holds K = {k0+8..k0+15, k0+24..k0+31} -> two contiguous b128 loads.
__device__ __forceinline__ v16h load_frag(const _Float16* __restrict__ base,
                                          int ld, int row0, int k0) {
  const int lane = threadIdx.x & 31;
  const int r  = row0 + (lane & 15);
  const int kb = k0 + ((lane >> 4) << 3);
  const _Float16* p = base + (size_t)r * (size_t)ld + kb;
  v8h lo = *reinterpret_cast<const v8h*>(p);
  v8h hi = *reinterpret_cast<const v8h*>(p + 16);
  v16h f;
#pragma unroll
  for (int i = 0; i < 8; ++i) { f[i] = lo[i]; f[i + 8] = hi[i]; }
  return f;
}

// ---------------- stage 0: precompute / transposes (f32 -> f16) -------------

__global__ void k_cvt_f16(const float* __restrict__ in, _Float16* __restrict__ out, int n) {
  int i = blockIdx.x * blockDim.x + threadIdx.x;
  if (i < n) out[i] = (_Float16)in[i];
}

// W [NHEADS][H][D] f32 -> Wt [NHEADS][D][H] f16
__global__ void k_transpose_w(const float* __restrict__ in, _Float16* __restrict__ out) {
  int i = blockIdx.x * blockDim.x + threadIdx.x;
  if (i >= NHEADS * HH * DD) return;
  int n   = i / (HH * DD);
  int rem = i - n * (HH * DD);
  int h   = rem / DD;
  int d   = rem - h * DD;
  out[(size_t)n * DD * HH + (size_t)d * HH + h] = (_Float16)in[i];
}

// Wo [H][H] f32 (in x out) -> Wot [H][H] f16 (out x in)
__global__ void k_transpose_wo(const float* __restrict__ in, _Float16* __restrict__ out) {
  int idx = blockIdx.x * blockDim.x + threadIdx.x;
  if (idx >= HH * HH) return;
  int i = idx / HH, o = idx - i * HH;
  out[(size_t)o * HH + i] = (_Float16)in[idx];
}

// ---------------- stage 1: fused QKV projection GEMM ------------------------
// grid (BL/16, NHEADS, 3), one wave per block; 16 tokens x 64 head-cols.
// 2x-unrolled ping-pong double buffer: no rotation moves, loads overlap WMMAs.
__global__ void __launch_bounds__(32) k_qkv(
    const _Float16* __restrict__ xh,
    const _Float16* __restrict__ wqt, const _Float16* __restrict__ wkt,
    const _Float16* __restrict__ wvt,
    const float* __restrict__ bq, const float* __restrict__ bk,
    const float* __restrict__ bv,
    _Float16* __restrict__ qh, _Float16* __restrict__ kh,
    _Float16* __restrict__ vt) {
  const int m0 = blockIdx.x * 16;
  const int n  = blockIdx.y;
  const int z  = blockIdx.z;
  const _Float16* wt   = (z == 0) ? wqt : (z == 1) ? wkt : wvt;
  const float*    bias = (z == 0) ? bq  : (z == 1) ? bk  : bv;
  const _Float16* wbase = wt + (size_t)n * DD * HH;

  v8f acc[4] = {v8f{}, v8f{}, v8f{}, v8f{}};
  v16h a0 = load_frag(xh, HH, m0, 0);
  v16h b0[4];
#pragma unroll
  for (int t = 0; t < 4; ++t) b0[t] = load_frag(wbase, HH, t * 16, 0);

  for (int k0 = 0; k0 < HH; k0 += 64) {
    // buffer 1 <- k0+32 (HH % 64 == 0 so always valid)
    v16h a1 = load_frag(xh, HH, m0, k0 + 32);
    v16h b1[4];
#pragma unroll
    for (int t = 0; t < 4; ++t) b1[t] = load_frag(wbase, HH, t * 16, k0 + 32);
#pragma unroll
    for (int t = 0; t < 4; ++t) acc[t] = wmma_f32_f16(a0, b0[t], acc[t]);
    // buffer 0 <- k0+64
    if (k0 + 64 < HH) {
      a0 = load_frag(xh, HH, m0, k0 + 64);
#pragma unroll
      for (int t = 0; t < 4; ++t) b0[t] = load_frag(wbase, HH, t * 16, k0 + 64);
    }
#pragma unroll
    for (int t = 0; t < 4; ++t) acc[t] = wmma_f32_f16(a1, b1[t], acc[t]);
  }

  const int lane = threadIdx.x & 31;
  const int nloc = lane & 15;
  const int mhi  = (lane >> 4) << 3;
  const int bidx = m0 >> 11;        // / L
  const int l0   = m0 & (LL - 1);

#pragma unroll
  for (int t = 0; t < 4; ++t) {
    const float bval = bias[n * DD + t * 16 + nloc];
#pragma unroll
    for (int j = 0; j < 8; ++j) acc[t][j] += bval;
  }

  if (z < 2) {  // Q/K: row-major [b][n][L][D]
    _Float16* outp = ((z == 0) ? qh : kh) + (size_t)(bidx * NHEADS + n) * LL * DD;
#pragma unroll
    for (int t = 0; t < 4; ++t) {
      const int col = t * 16 + nloc;
#pragma unroll
      for (int j = 0; j < 8; ++j) {
        const int row = l0 + mhi + j;
        outp[(size_t)row * DD + col] = (_Float16)acc[t][j];
      }
    }
  } else {      // V: transposed [b][n][D][L]  (B^T layout for P*V)
    _Float16* outp = vt + (size_t)(bidx * NHEADS + n) * DD * LL;
#pragma unroll
    for (int t = 0; t < 4; ++t) {
      const int col = t * 16 + nloc;
#pragma unroll
      for (int j = 0; j < 8; ++j) {
        const int row = l0 + mhi + j;
        outp[(size_t)col * LL + row] = (_Float16)acc[t][j];
      }
    }
  }
}

// ---------------- stage 2: fused masked flash attention ---------------------
// grid (L/32, NHEADS, B); one wave handles a 32-query tile over all 2048 keys.
// K fragments ping-pong double-buffered across 64-key macro-steps; V loads are
// issued before the softmax VALU block so exp/shuffle work hides their latency.
__global__ void __launch_bounds__(32) k_attn(
    const _Float16* __restrict__ qh, const _Float16* __restrict__ kh,
    const _Float16* __restrict__ vt, const int* __restrict__ mask,
    _Float16* __restrict__ cc) {
  __shared__ __align__(16) _Float16 plds[32 * 32];
  const int q0 = blockIdx.x * 32;
  const int n  = blockIdx.y;
  const int b  = blockIdx.z;
  const size_t ho = (size_t)(b * NHEADS + n);
  const _Float16* qptr = qh + ho * LL * DD;
  const _Float16* kptr = kh + ho * LL * DD;
  const _Float16* vptr = vt + ho * DD * LL;
  const int* mptr = mask + b * LL;

  const int lane = threadIdx.x & 31;
  const int nloc = lane & 15;
  const int mhi  = (lane >> 4) << 3;
  const float scale = 0.125f;  // 1/sqrt(64)

  // Q fragments: 2 query sub-tiles x 2 d-chunks
  v16h aq[2][2];
#pragma unroll
  for (int i = 0; i < 2; ++i) {
    aq[i][0] = load_frag(qptr, DD, q0 + i * 16, 0);
    aq[i][1] = load_frag(qptr, DD, q0 + i * 16, 32);
  }

  float mst[2][8], lst[2][8];
#pragma unroll
  for (int i = 0; i < 2; ++i)
#pragma unroll
    for (int j = 0; j < 8; ++j) { mst[i][j] = -INFINITY; lst[i][j] = 0.0f; }
  v8f o[2][4];
#pragma unroll
  for (int i = 0; i < 2; ++i)
#pragma unroll
    for (int t = 0; t < 4; ++t) o[i][t] = v8f{};

  // process one 32-key chunk with the given K fragments (inlined twice)
  auto chunk = [&](int kc, v16h (&bk)[2][2]) {
    // prefetch K/V two chunks ahead into the cache hierarchy
    if (kc + 64 < LL) {
      __builtin_prefetch(kptr + (size_t)(kc + 64 + lane) * DD, 0, 1);
      __builtin_prefetch(vptr + (size_t)lane * LL + kc + 64, 0, 1);
      __builtin_prefetch(vptr + (size_t)(lane + 32) * LL + kc + 64, 0, 1);
    }
    // scores: 2 query sub-tiles x 2 key sub-tiles, contraction over d=64
    v8f s[2][2];
#pragma unroll
    for (int i = 0; i < 2; ++i)
#pragma unroll
      for (int jt = 0; jt < 2; ++jt) {
        v8f sv = {};
        sv = wmma_f32_f16(aq[i][0], bk[jt][0], sv);
        sv = wmma_f32_f16(aq[i][1], bk[jt][1], sv);
        s[i][jt] = sv;
      }

    // V fragments for this chunk (in flight while softmax VALU runs)
    v16h bvf[4];
#pragma unroll
    for (int t = 0; t < 4; ++t) bvf[t] = load_frag(vptr, LL, t * 16, kc);

    const bool vk0 = mptr[kc + nloc] != 0;
    const bool vk1 = mptr[kc + 16 + nloc] != 0;

    float p[2][2][8], alpha[2][8];
#pragma unroll
    for (int i = 0; i < 2; ++i) {
#pragma unroll
      for (int j = 0; j < 8; ++j) {
        p[i][0][j] = vk0 ? s[i][0][j] * scale : -1e30f;
        p[i][1][j] = vk1 ? s[i][1][j] * scale : -1e30f;
      }
      // online softmax: row max / row sum via 16-lane butterflies
#pragma unroll
      for (int j = 0; j < 8; ++j) {
        float cm = fmaxf(p[i][0][j], p[i][1][j]);
        cm = fmaxf(cm, __shfl_xor(cm, 1, 32));
        cm = fmaxf(cm, __shfl_xor(cm, 2, 32));
        cm = fmaxf(cm, __shfl_xor(cm, 4, 32));
        cm = fmaxf(cm, __shfl_xor(cm, 8, 32));
        const float nm = fmaxf(mst[i][j], cm);
        alpha[i][j] = __expf(mst[i][j] - nm);
        mst[i][j] = nm;
        p[i][0][j] = __expf(p[i][0][j] - nm);
        p[i][1][j] = __expf(p[i][1][j] - nm);
        float rs = p[i][0][j] + p[i][1][j];
        rs += __shfl_xor(rs, 1, 32);
        rs += __shfl_xor(rs, 2, 32);
        rs += __shfl_xor(rs, 4, 32);
        rs += __shfl_xor(rs, 8, 32);
        lst[i][j] = lst[i][j] * alpha[i][j] + rs;
      }
#pragma unroll
      for (int t = 0; t < 4; ++t)
#pragma unroll
        for (int j = 0; j < 8; ++j) o[i][t][j] *= alpha[i][j];
    }

    // C-layout -> A-layout for P via LDS bounce (32x32 f16 tile)
    __syncthreads();
#pragma unroll
    for (int i = 0; i < 2; ++i)
#pragma unroll
      for (int j = 0; j < 8; ++j) {
        const int row = i * 16 + mhi + j;
        plds[row * 32 + nloc]      = (_Float16)p[i][0][j];
        plds[row * 32 + 16 + nloc] = (_Float16)p[i][1][j];
      }
    __syncthreads();
    const int r  = lane & 15;
    const int kb = (lane >> 4) << 3;
    v16h ap[2];
#pragma unroll
    for (int i = 0; i < 2; ++i) {
      v8h lo = *reinterpret_cast<const v8h*>(&plds[(i * 16 + r) * 32 + kb]);
      v8h hi = *reinterpret_cast<const v8h*>(&plds[(i * 16 + r) * 32 + kb + 16]);
#pragma unroll
      for (int e = 0; e < 8; ++e) { ap[i][e] = lo[e]; ap[i][e + 8] = hi[e]; }
    }

    // O += P(32x32) * V(32x64); Vt rows are B^T rows; V frags shared by both i
#pragma unroll
    for (int t = 0; t < 4; ++t) {
      o[0][t] = wmma_f32_f16(ap[0], bvf[t], o[0][t]);
      o[1][t] = wmma_f32_f16(ap[1], bvf[t], o[1][t]);
    }
  };

  // ping-pong K double buffer over 64-key macro-iterations
  v16h bk0[2][2], bk1[2][2];
#pragma unroll
  for (int jt = 0; jt < 2; ++jt) {
    bk0[jt][0] = load_frag(kptr, DD, jt * 16, 0);
    bk0[jt][1] = load_frag(kptr, DD, jt * 16, 32);
  }
  for (int kc = 0; kc < LL; kc += 64) {
    // buffer 1 <- keys kc+32 (LL % 64 == 0 so always valid)
#pragma unroll
    for (int jt = 0; jt < 2; ++jt) {
      bk1[jt][0] = load_frag(kptr, DD, kc + 32 + jt * 16, 0);
      bk1[jt][1] = load_frag(kptr, DD, kc + 32 + jt * 16, 32);
    }
    chunk(kc, bk0);
    // buffer 0 <- keys kc+64
    if (kc + 64 < LL) {
#pragma unroll
      for (int jt = 0; jt < 2; ++jt) {
        bk0[jt][0] = load_frag(kptr, DD, kc + 64 + jt * 16, 0);
        bk0[jt][1] = load_frag(kptr, DD, kc + 64 + jt * 16, 32);
      }
    }
    chunk(kc + 32, bk1);
  }

  // normalize and emit concat [b][L][H] f16 with head offset
  _Float16* outp = cc + (size_t)b * LL * HH;
#pragma unroll
  for (int i = 0; i < 2; ++i) {
    float inv[8];
#pragma unroll
    for (int j = 0; j < 8; ++j) inv[j] = 1.0f / lst[i][j];
#pragma unroll
    for (int t = 0; t < 4; ++t) {
      const int col = n * DD + t * 16 + nloc;
#pragma unroll
      for (int j = 0; j < 8; ++j) {
        const int row = q0 + i * 16 + mhi + j;
        outp[(size_t)row * HH + col] = (_Float16)(o[i][t][j] * inv[j]);
      }
    }
  }
}

// ---------------- stage 3: output projection --------------------------------
// grid (BL/16, H/64); 16 rows x 64 cols per wave; 2x-unrolled ping-pong.
__global__ void __launch_bounds__(32) k_out(
    const _Float16* __restrict__ cc, const _Float16* __restrict__ wot,
    const float* __restrict__ bo, float* __restrict__ out) {
  const int m0 = blockIdx.x * 16;
  const int c0 = blockIdx.y * 64;
  v8f acc[4] = {v8f{}, v8f{}, v8f{}, v8f{}};
  v16h a0 = load_frag(cc, HH, m0, 0);
  v16h b0[4];
#pragma unroll
  for (int t = 0; t < 4; ++t) b0[t] = load_frag(wot, HH, c0 + t * 16, 0);

  for (int k0 = 0; k0 < HH; k0 += 64) {
    v16h a1 = load_frag(cc, HH, m0, k0 + 32);
    v16h b1[4];
#pragma unroll
    for (int t = 0; t < 4; ++t) b1[t] = load_frag(wot, HH, c0 + t * 16, k0 + 32);
#pragma unroll
    for (int t = 0; t < 4; ++t) acc[t] = wmma_f32_f16(a0, b0[t], acc[t]);
    if (k0 + 64 < HH) {
      a0 = load_frag(cc, HH, m0, k0 + 64);
#pragma unroll
      for (int t = 0; t < 4; ++t) b0[t] = load_frag(wot, HH, c0 + t * 16, k0 + 64);
    }
#pragma unroll
    for (int t = 0; t < 4; ++t) acc[t] = wmma_f32_f16(a1, b1[t], acc[t]);
  }

  const int lane = threadIdx.x & 31;
  const int nloc = lane & 15;
  const int mhi  = (lane >> 4) << 3;
#pragma unroll
  for (int t = 0; t < 4; ++t) {
    const int col = c0 + t * 16 + nloc;
    const float bval = bo[col];
#pragma unroll
    for (int j = 0; j < 8; ++j) {
      const int row = m0 + mhi + j;
      out[(size_t)row * HH + col] = acc[t][j] + bval;
    }
  }
}

// ---------------- host launcher ---------------------------------------------

extern "C" void kernel_launch(void* const* d_in, const int* in_sizes, int n_in,
                              void* d_out, int out_size, void* d_ws, size_t ws_size,
                              hipStream_t stream) {
  const float* x    = (const float*)d_in[0];
  const float* Wq   = (const float*)d_in[1];
  const float* bq   = (const float*)d_in[2];
  const float* Wk   = (const float*)d_in[3];
  const float* bk   = (const float*)d_in[4];
  const float* Wv   = (const float*)d_in[5];
  const float* bv   = (const float*)d_in[6];
  const float* Wo   = (const float*)d_in[7];
  const float* bo   = (const float*)d_in[8];
  const int*   mask = (const int*)d_in[9];
  float* out = (float*)d_out;

  // workspace layout (f16 elements)
  _Float16* ws = (_Float16*)d_ws;
  const size_t SZ_X  = (size_t)BL * HH;           // 8 Mi
  const size_t SZ_W  = (size_t)NHEADS * DD * HH;  // 1 Mi
  const size_t SZ_WO = (size_t)HH * HH;           // 1 Mi
  const size_t SZ_QK = (size_t)BB * NHEADS * LL * DD;  // 8 Mi
  _Float16* xh  = ws;
  _Float16* wqt = xh  + SZ_X;
  _Float16* wkt = wqt + SZ_W;
  _Float16* wvt = wkt + SZ_W;
  _Float16* wot = wvt + SZ_W;
  _Float16* qh  = wot + SZ_WO;
  _Float16* kh  = qh  + SZ_QK;
  _Float16* vt  = kh  + SZ_QK;
  _Float16* cc  = vt  + SZ_QK;   // total ~46.1 Mi f16 = ~92.3 MB

  // stage 0: conversions / transposes
  k_cvt_f16<<<(BL * HH + 255) / 256, 256, 0, stream>>>(x, xh, BL * HH);
  const int wthreads = NHEADS * HH * DD;
  k_transpose_w<<<(wthreads + 255) / 256, 256, 0, stream>>>(Wq, wqt);
  k_transpose_w<<<(wthreads + 255) / 256, 256, 0, stream>>>(Wk, wkt);
  k_transpose_w<<<(wthreads + 255) / 256, 256, 0, stream>>>(Wv, wvt);
  k_transpose_wo<<<(HH * HH + 255) / 256, 256, 0, stream>>>(Wo, wot);

  // stage 1: QKV projections
  k_qkv<<<dim3(BL / 16, NHEADS, 3), 32, 0, stream>>>(
      xh, wqt, wkt, wvt, bq, bk, bv, qh, kh, vt);

  // stage 2: fused masked-softmax attention (32-query tiles)
  k_attn<<<dim3(LL / 32, NHEADS, BB), 32, 0, stream>>>(qh, kh, vt, mask, cc);

  // stage 3: output projection
  k_out<<<dim3(BL / 16, HH / 64), 32, 0, stream>>>(cc, wot, bo, out);
}
